// GraphAttentionLayer_16527034155244
// MI455X (gfx1250) — compile-verified, matrix operand order checked by reading, not run on hardware
//
#include <hip/hip_runtime.h>
#include <hip/hip_bf16.h>

// ---------------------------------------------------------------------------
// GAT layer, fused for MI455X (gfx1250, wave32, WMMA bf16 16x16x32).
// Stage 0: W (128x512 f32) -> bf16
// Stage 1: h = X@W^T + b  (WMMA), also s1 = h@a1, s2 = h@a2, store hT (bf16)
// Stage 2: fused masked softmax-attention, j-split 4x across a 4-wave
//          workgroup (2048 waves total) with LDS partial-accumulator combine.
// ---------------------------------------------------------------------------

typedef __attribute__((ext_vector_type(16))) __bf16 v16bf;
typedef __attribute__((ext_vector_type(8)))  __bf16 v8bf;
typedef __attribute__((ext_vector_type(8)))  float  v8f;
typedef __attribute__((ext_vector_type(4)))  float  v4f;
typedef __attribute__((ext_vector_type(4)))  int    v4i;

#define N_NODES 8192
#define F_IN    512
#define F_OUT   128
#define JSEG    (N_NODES / 4)   // 2048 columns per wave segment
#define LDS_STRIDE 68           // 64 payload floats + 4 pad (bank decorrelate)

// ---------------- Stage 0: convert W_w to bf16 ------------------------------
__global__ __launch_bounds__(256) void gat_cvtw_kernel(
    const float* __restrict__ w, __bf16* __restrict__ wb) {
  int i = blockIdx.x * 256 + threadIdx.x;   // 128*512 = 65536 elements
  wb[i] = (__bf16)w[i];
}

// ---------------- Stage 1: h, s1, s2, hT ------------------------------------
// One wave handles a 16-row block of X, producing h[16][128] via 8 WMMA accs.
__global__ __launch_bounds__(256) void gat_h_kernel(
    const float*  __restrict__ X,
    const __bf16* __restrict__ Wbf,   // [128][512] bf16 (same layout as W_w)
    const float*  __restrict__ Wb,    // [128]
    const float*  __restrict__ aw,    // [256] = a1 ++ a2
    __bf16*       __restrict__ hT,    // [128][8192] bf16 (transposed h)
    float*        __restrict__ s1g,   // [8192]
    float*        __restrict__ s2g) { // [8192]
  const int lane = threadIdx.x & 31;
  const int wid  = (blockIdx.x * blockDim.x + threadIdx.x) >> 5;
  const int sel  = lane >> 4;     // which 16-lane half (K-half select)
  const int r    = lane & 15;     // A: row M; B/D: column N within tile
  const int i0   = wid * 16;

  v8f acc[8];
  const v8f vzero = {0.f, 0.f, 0.f, 0.f, 0.f, 0.f, 0.f, 0.f};
#pragma unroll
  for (int t = 0; t < 8; ++t) acc[t] = vzero;

  const float* xrow = X + (size_t)(i0 + r) * F_IN;

  for (int k0 = 0; k0 < F_IN; k0 += 32) {
    // A fragment: 16x32 bf16. Lane half sel: K = 8*sel + e  (e<8)
    //                                        K = 16 + 8*sel + (e-8) (e>=8)
    const float* xp = xrow + k0 + 8 * sel;
    v4f x0 = *(const v4f*)(xp);
    v4f x1 = *(const v4f*)(xp + 4);
    v4f x2 = *(const v4f*)(xp + 16);
    v4f x3 = *(const v4f*)(xp + 20);
    v16bf a;
#pragma unroll
    for (int e = 0; e < 4; ++e) {
      a[e]      = (__bf16)x0[e];
      a[e + 4]  = (__bf16)x1[e];
      a[e + 8]  = (__bf16)x2[e];
      a[e + 12] = (__bf16)x3[e];
    }
#pragma unroll
    for (int t = 0; t < 8; ++t) {
      // B fragment: B[k][n] = W[n][k], lane supplies column n = 16t + r.
      const __bf16* wp = Wbf + (size_t)(16 * t + r) * F_IN + k0 + 8 * sel;
      v8bf b0 = *(const v8bf*)(wp);
      v8bf b1 = *(const v8bf*)(wp + 16);
      v16bf b;
#pragma unroll
      for (int e = 0; e < 8; ++e) { b[e] = b0[e]; b[e + 8] = b1[e]; }
      acc[t] = __builtin_amdgcn_wmma_f32_16x16x32_bf16(
          false, a, false, b, (short)0, acc[t], false, false);
    }
  }

  // Epilogue: bias, s1/s2 row dots, hT store.
  // D layout: lane holds column n = 16t + r, rows m = 8*sel + v (v = VGPR idx).
  float sp1[8], sp2[8];
#pragma unroll
  for (int v = 0; v < 8; ++v) { sp1[v] = 0.f; sp2[v] = 0.f; }

#pragma unroll
  for (int t = 0; t < 8; ++t) {
    const int n = 16 * t + r;
    const float bias = Wb[n];
    const float a1v  = aw[n];
    const float a2v  = aw[F_OUT + n];
    v8bf hv;
#pragma unroll
    for (int v = 0; v < 8; ++v) {
      float h = acc[t][v] + bias;
      sp1[v] += h * a1v;
      sp2[v] += h * a2v;
      hv[v] = (__bf16)h;
    }
    // hT[n][i0 + 8*sel + 0..7]  -- 8 consecutive bf16 = one b128 store
    *(v8bf*)(hT + (size_t)n * N_NODES + i0 + 8 * sel) = hv;
  }

  // Reduce s1/s2 partials across the 16 columns (stay within each lane-half).
#pragma unroll
  for (int m = 1; m < 16; m <<= 1) {
#pragma unroll
    for (int v = 0; v < 8; ++v) {
      sp1[v] += __shfl_xor(sp1[v], m, 32);
      sp2[v] += __shfl_xor(sp2[v], m, 32);
    }
  }
  if (r == 0) {
#pragma unroll
    for (int v = 0; v < 8; ++v) {
      s1g[i0 + 8 * sel + v] = sp1[v];
      s2g[i0 + 8 * sel + v] = sp2[v];
    }
  }
}

// ---------------- Stage 2: fused masked softmax + attention@h ---------------
// Workgroup = 128 threads = 4 waves = one 16-row block; wave w covers
// j in [w*2048, w*2048+2048). relu => exponent bounded => single pass, no
// running-max rescale: masked p=0, unmasked p=exp(relu(.)) in [1, e^~10].
// Waves 1..3 dump partial accumulators/rowsums to LDS; wave 0 combines.
__global__ __launch_bounds__(128) void gat_attn_kernel(
    const int*    __restrict__ adj,   // [8192][8192] int32 (0/1)
    const __bf16* __restrict__ hT,    // [128][8192]
    const float*  __restrict__ s1g,
    const float*  __restrict__ s2g,
    const float*  __restrict__ abp,   // [1]
    float*        __restrict__ out) { // [8192][128]
  __shared__ float red[3 * 32 * LDS_STRIDE];  // partial 16x128 tiles, waves 1..3
  __shared__ float rsumLds[4 * 16];           // partial row sums, all 4 waves

  const int lane = threadIdx.x & 31;
  const int w    = threadIdx.x >> 5;  // j-segment 0..3
  const int sel  = lane >> 4;
  const int r    = lane & 15;
  const int i0   = blockIdx.x * 16;

  const int* arow = adj + (size_t)(i0 + r) * N_NODES;
  const float base = s1g[i0 + r] + abp[0];

  v8f acc[8];
  const v8f vzero = {0.f, 0.f, 0.f, 0.f, 0.f, 0.f, 0.f, 0.f};
#pragma unroll
  for (int t = 0; t < 8; ++t) acc[t] = vzero;
  float rowsum = 0.f;

  const int jlo = w * JSEG;
  const int jhi = jlo + JSEG;
  for (int j0 = jlo; j0 < jhi; j0 += 32) {
    const int jb1 = j0 + 8 * sel;       // K = 0..7   (this half)
    const int jb2 = jb1 + 16;           // K = 16..23 (this half)
    // adj tile: 16 ints per lane, two contiguous 32B runs
    v4i ad0 = *(const v4i*)(arow + jb1);
    v4i ad1 = *(const v4i*)(arow + jb1 + 4);
    v4i ad2 = *(const v4i*)(arow + jb2);
    v4i ad3 = *(const v4i*)(arow + jb2 + 4);
    // matching s2 values
    v4f t0 = *(const v4f*)(s2g + jb1);
    v4f t1 = *(const v4f*)(s2g + jb1 + 4);
    v4f t2 = *(const v4f*)(s2g + jb2);
    v4f t3 = *(const v4f*)(s2g + jb2 + 4);

    // prefetch next adj tile into cache (global_prefetch_b8)
    __builtin_prefetch(arow + jb1 + 32, 0, 1);

    v16bf a;
#pragma unroll
    for (int e = 0; e < 4; ++e) {
      float p0 = (ad0[e] > 0) ? __expf(fmaxf(base + t0[e], 0.f)) : 0.f;
      float p1 = (ad1[e] > 0) ? __expf(fmaxf(base + t1[e], 0.f)) : 0.f;
      float p2 = (ad2[e] > 0) ? __expf(fmaxf(base + t2[e], 0.f)) : 0.f;
      float p3 = (ad3[e] > 0) ? __expf(fmaxf(base + t3[e], 0.f)) : 0.f;
      rowsum += (p0 + p1) + (p2 + p3);
      a[e]      = (__bf16)p0;
      a[e + 4]  = (__bf16)p1;
      a[e + 8]  = (__bf16)p2;
      a[e + 12] = (__bf16)p3;
    }

#pragma unroll
    for (int t = 0; t < 8; ++t) {
      // B[k][n] = h[j0+k][16t+n]; lane supplies column n = r -> hT row 16t+r,
      // contiguous in j: two 16B loads.
      const __bf16* hp = hT + (size_t)(16 * t + r) * N_NODES + jb1;
      v8bf b0 = *(const v8bf*)(hp);
      v8bf b1 = *(const v8bf*)(hp + 16);
      v16bf b;
#pragma unroll
      for (int e = 0; e < 8; ++e) { b[e] = b0[e]; b[e + 8] = b1[e]; }
      acc[t] = __builtin_amdgcn_wmma_f32_16x16x32_bf16(
          false, a, false, b, (short)0, acc[t], false, false);
    }
  }

  // Segment row sums: lanes l and l+16 hold complementary K-halves of row r.
  rowsum += __shfl_xor(rowsum, 16, 32);
  if (sel == 0) rsumLds[w * 16 + r] = rowsum;

  // Waves 1..3: publish 16x128 partial accumulator tile to LDS.
  if (w != 0) {
    float* dst = red + ((w - 1) * 32 + lane) * LDS_STRIDE;
#pragma unroll
    for (int t = 0; t < 8; ++t) {
      v4f lo, hi;
#pragma unroll
      for (int e = 0; e < 4; ++e) { lo[e] = acc[t][e]; hi[e] = acc[t][e + 4]; }
      *(v4f*)(dst + t * 8)     = lo;
      *(v4f*)(dst + t * 8 + 4) = hi;
    }
  }
  __syncthreads();

  if (w == 0) {
    // Combine the 3 partial tiles into registers.
#pragma unroll
    for (int s = 0; s < 3; ++s) {
      const float* src = red + (s * 32 + lane) * LDS_STRIDE;
#pragma unroll
      for (int t = 0; t < 8; ++t) {
        v4f lo = *(const v4f*)(src + t * 8);
        v4f hi = *(const v4f*)(src + t * 8 + 4);
        v8f add;
#pragma unroll
        for (int e = 0; e < 4; ++e) { add[e] = lo[e]; add[e + 4] = hi[e]; }
        acc[t] += add;
      }
    }
    // Full row sums + normalize. Rows m = 8*sel + v for this lane.
    float inv[8];
#pragma unroll
    for (int v = 0; v < 8; ++v) {
      const int m = 8 * sel + v;
      float rs = rsumLds[m] + rsumLds[16 + m] + rsumLds[32 + m] + rsumLds[48 + m];
      inv[v] = 1.0f / fmaxf(rs, 1e-30f);   // unmasked rows have rowsum >= 1
    }
#pragma unroll
    for (int t = 0; t < 8; ++t) {
#pragma unroll
      for (int v = 0; v < 8; ++v) {
        out[(size_t)(i0 + 8 * sel + v) * F_OUT + 16 * t + r] = acc[t][v] * inv[v];
      }
    }
  }
}

// ---------------------------------------------------------------------------
extern "C" void kernel_launch(void* const* d_in, const int* in_sizes, int n_in,
                              void* d_out, int out_size, void* d_ws, size_t ws_size,
                              hipStream_t stream) {
  const float* X   = (const float*)d_in[0];   // [8192][512]
  const int*   adj = (const int*)  d_in[1];   // [8192][8192]
  const float* Ww  = (const float*)d_in[2];   // [128][512]
  const float* Wb  = (const float*)d_in[3];   // [128]
  const float* aw  = (const float*)d_in[4];   // [256]
  const float* ab  = (const float*)d_in[5];   // [1]
  float* out = (float*)d_out;                 // [8192][128]

  // Workspace layout (~2.25 MB total):
  char* ws = (char*)d_ws;
  __bf16* hT  = (__bf16*)ws;                              // 2 MB
  float*  s1  = (float*)(ws + (size_t)2 * 1024 * 1024);   // 32 KB
  float*  s2  = s1 + N_NODES;                             // 32 KB
  __bf16* Wbf = (__bf16*)(s2 + N_NODES);                  // 128 KB

  gat_cvtw_kernel<<<dim3((F_OUT * F_IN) / 256), dim3(256), 0, stream>>>(Ww, Wbf);
  gat_h_kernel<<<dim3(N_NODES / 16 / 8), dim3(256), 0, stream>>>(
      X, Wbf, Wb, aw, hT, s1, s2);
  gat_attn_kernel<<<dim3(N_NODES / 16), dim3(128), 0, stream>>>(
      adj, hT, s1, s2, ab, out);
}
